// OneHotGRULM_88321707475242
// MI455X (gfx1250) — matrix operand order repaired and checked by reading, not compiled, not run on hardware
//
#include <hip/hip_runtime.h>
#include <hip/hip_bf16.h>

// ---------------------------------------------------------------------------
// OneHotGRULM on MI455X (gfx1250, wave32, WMMA + TDM).
//
//   1. init:    h ping-pong state + barrier counters.
//   2. pack:    w_hh_0/1, w_ih_1, head_w  f32 -> f16 WMMA-B fragment layout.
//   3. gather:  gi0 = w_ih_0[:, idx] + b_ih_0  (one-hot GEMM == gather).
//   4. scan L0: 64 single-wave persistent blocks; per-block w_hh slice (96 KB)
//               preloaded into LDS; 3 WMMAs/k-tile off one A fragment;
//               device-scope split barrier per step.
//   5. GEMM:    gi1 = out0 x w_ih_1^T + b_ih_1.
//   6. scan L1.
//   7. GEMM:    logits = out1 x head_w^T + head_b  (268 GFLOP roofline kernel).
//
// GEMM: 4 waves/block, block tile 64x256, A (128 KB) staged into LDS via the
// Tensor Data Mover (tensor_load_to_lds + s_wait_tensorcnt), 16 WMMAs per
// k-tile per wave (4x4 register tile).
// ---------------------------------------------------------------------------

typedef __attribute__((ext_vector_type(8)))  _Float16 v8h;
typedef __attribute__((ext_vector_type(16))) _Float16 v16h;
typedef __attribute__((ext_vector_type(8)))  float    v8f;
typedef __attribute__((ext_vector_type(4)))  unsigned u32x4;
typedef __attribute__((ext_vector_type(8)))  int      i32x8;
typedef __attribute__((ext_vector_type(4)))  int      i32x4;

#define VV   32000
#define HH   1024
#define BB   8
#define TT   512
#define G3H  3072
#define KT   32        // k-tiles of 32 along H (1024/32)
#define SCAN_NB 64     // single-wave persistent blocks (64 x 16 cols = 1024)

// ---- WMMA fragment loaders -------------------------------------------------
// A (16x32 f16, MxK): lane l<16 -> row M=l,   K {0..7,16..23}
//                     lane l>=16 -> row M=l-16, K {8..15,24..31}
__device__ __forceinline__ v16h load_a_frag(const _Float16* rowptr, int kt, int lane) {
  const _Float16* p = rowptr + kt * 32 + ((lane & 16) ? 8 : 0);
  v8h lo = *(const v8h*)p;
  v8h hi = *(const v8h*)(p + 16);
  return __builtin_shufflevector(lo, hi, 0, 1, 2, 3, 4, 5, 6, 7,
                                 8, 9, 10, 11, 12, 13, 14, 15);
}

// B (32x16 f16, KxN) pre-packed: tile (nt,kt) = 32 lanes x 16 contiguous halves.
__device__ __forceinline__ v16h load_b_frag(const _Float16* pk, int nt, int kt, int lane) {
  const _Float16* p = pk + (((size_t)nt * KT + kt) * 32 + (size_t)lane) * 16;
  return *(const v16h*)p;
}

__device__ __forceinline__ v8f wmma_f16(v16h a, v16h b, v8f c) {
  return __builtin_amdgcn_wmma_f32_16x16x32_f16(false, a, false, b, (short)0, c,
                                                false, false);
}

__device__ __forceinline__ float sigmoidf_(float x) {
  return 1.0f / (1.0f + __expf(-x));
}

// ---- TDM: DMA a [rows x 1024] f16 row-major tile from global into LDS ------
// D# built per CDNA5 ISA ch.8 (2-D tensor, data_size=2B). 6-arg builtin form
// (clang-23 / therock-10.0): (g0, g1, g2, g3, g4, cpol).
__device__ __forceinline__ void tdm_load_tile_f16(const _Float16* gsrc,
                                                  unsigned lds_byte_addr,
                                                  unsigned rows) {
  unsigned long long ga = (unsigned long long)(size_t)gsrc;
  u32x4 g0;
  g0[0] = 1u;                                    // count=1, user mode
  g0[1] = lds_byte_addr;                         // lds_addr
  g0[2] = (unsigned)(ga & 0xffffffffu);          // global_addr[31:0]
  g0[3] = (unsigned)((ga >> 32) & 0x01ffffffu)   // global_addr[56:32]
          | (2u << 30);                          // type = 2 ("image")
  i32x8 g1;
  g1[0] = (int)(1u << 16);                       // data_size=1 -> 2 bytes
  g1[1] = (int)(1024u << 16);                    // tensor_dim0 = 1024
  g1[2] = (int)(65535u << 16);                   // tensor_dim1 (no OOB clip)
  g1[3] = (int)(1024u << 16);                    // tile_dim0 = 1024
  g1[4] = (int)rows;                             // tile_dim1 = rows, tile_dim2=0
  g1[5] = 1024;                                  // tensor_dim0_stride = 1024
  g1[6] = 0;
  g1[7] = 0;
  i32x4 z4 = {0, 0, 0, 0};
  i32x8 z8 = {0, 0, 0, 0, 0, 0, 0, 0};
  __builtin_amdgcn_tensor_load_to_lds(g0, g1, z4, z4, z8, 0);
}

// ---- 1. init ---------------------------------------------------------------
__global__ void init_kernel(const float* __restrict__ h0,
                            _Float16* __restrict__ h16,   // [2][2][16*HH]
                            float* __restrict__ h32,      // [2][2][BB*HH]
                            unsigned* __restrict__ ctr) { // [2]
  int tid = blockIdx.x * blockDim.x + threadIdx.x;
  int nthr = gridDim.x * blockDim.x;
  if (tid < 2) ctr[tid] = 0u;
  for (int i = tid; i < 2 * 2 * 16 * HH; i += nthr) {
    int layer = i / (2 * 16 * HH);
    int rem   = i % (2 * 16 * HH);
    int buf   = rem / (16 * HH);
    int rc    = rem % (16 * HH);
    int row = rc / HH, cc = rc % HH;
    float v = (buf == 0 && row < BB) ? h0[layer * BB * HH + row * HH + cc] : 0.0f;
    h16[i] = (_Float16)v;
  }
  for (int i = tid; i < 2 * 2 * BB * HH; i += nthr) {
    int layer = i / (2 * BB * HH);
    int rem   = i % (2 * BB * HH);
    int buf   = rem / (BB * HH);
    int rc    = rem % (BB * HH);
    h32[i] = (buf == 0) ? h0[layer * BB * HH + rc] : 0.0f;
  }
}

// ---- 2. pack f32 [N,K=HH] row-major -> WMMA-B fragment f16 layout ----------
__global__ void pack_b_kernel(const float* __restrict__ w,
                              _Float16* __restrict__ pk, int ntiles) {
  size_t slot  = (size_t)blockIdx.x * blockDim.x + threadIdx.x;
  size_t total = (size_t)ntiles * KT * 32;
  if (slot >= total) return;
  int    lane = (int)(slot & 31);
  size_t tile = slot >> 5;
  int    kt   = (int)(tile % KT);
  size_t nt   = tile / KT;
  size_t row  = nt * 16 + (lane & 15);
  int    ks   = kt * 32 + ((lane & 16) ? 8 : 0);
  const float* src = w + row * HH + ks;
  _Float16*    dst = pk + slot * 16;
#pragma unroll
  for (int i = 0; i < 16; ++i) dst[i] = (_Float16)src[i + (i & 8)];
}

// ---- 3. one-hot "GEMM" == column gather of w_ih_0 --------------------------
__global__ void gather_gi0_kernel(const int* __restrict__ idx,
                                  const float* __restrict__ w_ih0,
                                  const float* __restrict__ b_ih0,
                                  float* __restrict__ gi0) {
  int tok = blockIdx.x;
  int col = idx[tok];
  float* out = gi0 + (size_t)tok * G3H;
  for (int g = threadIdx.x; g < G3H; g += blockDim.x)
    out[g] = w_ih0[(size_t)g * VV + col] + b_ih0[g];
}

// ---- 4/6. persistent GRU scan ----------------------------------------------
// 64 blocks x 1 wave. Block owns 16 hidden cols; its w_hh slice (3 gates x
// 16 x 1024 f16 = 96 KB) lives in LDS for all 512 steps. Per k-tile: one A
// fragment (h, from global/L2) feeds 3 WMMAs whose B fragments come from LDS.
__global__ void __launch_bounds__(32)
gru_scan_kernel(const float* __restrict__ gi,      // [B*T, 3H] f32
                const _Float16* __restrict__ wpk,  // packed w_hh [3H, H]
                const float* __restrict__ b_hh,    // [3H]
                _Float16* __restrict__ out16,      // [B*T, H] f16
                _Float16* __restrict__ h16,        // [2][16*HH] ping-pong
                float* __restrict__ h32,           // [2][BB*HH] ping-pong
                unsigned* __restrict__ ctr,
                float* __restrict__ dout_h) {      // [BB*HH] final h (f32)
  extern __shared__ _Float16 wlds[];               // 96 KB: [3][32kt][32lane][16]
  const int lane = threadIdx.x;
  const int nt_r = blockIdx.x;                     // this block's r-gate ntile
  const int jcol = nt_r * 16 + (lane & 15);
  const bool act = lane < 16;

  // one-time preload of the w_hh slice into LDS (3 x 16384 halves)
  for (int c = lane; c < 6144; c += 32) {          // v8h chunks
    int g = c >> 11;                               // 0..2 (gate)
    int r = c & 2047;
    size_t nt = (size_t)(nt_r + g * (HH >> 4));
    *(v8h*)&wlds[(g << 14) + r * 8] = *(const v8h*)&wpk[nt * KT * 512 + (size_t)r * 8];
  }

  const float br = b_hh[jcol];
  const float bz = b_hh[HH + jcol];
  const float bn = b_hh[2 * HH + jcol];

  unsigned phase = 0;
  for (int t = 0; t < TT; ++t) {
    const _Float16* hin    = h16 + (size_t)(t & 1) * (16 * HH);
    _Float16*       hout   = h16 + (size_t)((t + 1) & 1) * (16 * HH);
    const float*    h32in  = h32 + (size_t)(t & 1) * (BB * HH);
    float*          h32out = h32 + (size_t)((t + 1) & 1) * (BB * HH);

    v8f cr = {}, cz = {}, cn = {};
    const _Float16* arow = hin + (size_t)(lane & 15) * HH;
#pragma unroll 4
    for (int kt = 0; kt < KT; ++kt) {
      v16h a = load_a_frag(arow, kt, lane);                 // global (L2) h
      v16h wr = *(const v16h*)&wlds[((0 * KT + kt) * 32 + lane) * 16];  // LDS
      v16h wz = *(const v16h*)&wlds[((1 * KT + kt) * 32 + lane) * 16];
      v16h wn = *(const v16h*)&wlds[((2 * KT + kt) * 32 + lane) * 16];
      cr = wmma_f16(a, wr, cr);
      cz = wmma_f16(a, wz, cz);
      cn = wmma_f16(a, wn, cn);
    }

    if (act) {
#pragma unroll
      for (int v = 0; v < BB; ++v) {
        const float* gp = gi + ((size_t)v * TT + t) * G3H + jcol;
        float i_r = gp[0], i_z = gp[HH], i_n = gp[2 * HH];
        float hold = h32in[v * HH + jcol];
        float r = sigmoidf_(i_r + cr[v] + br);
        float z = sigmoidf_(i_z + cz[v] + bz);
        float n = tanhf(i_n + r * (cn[v] + bn));
        float hnew = (1.0f - z) * n + z * hold;
        h32out[v * HH + jcol] = hnew;
        _Float16 hhalf = (_Float16)hnew;
        hout[v * HH + jcol] = hhalf;
        out16[((size_t)v * TT + t) * HH + jcol] = hhalf;
        if (t == TT - 1) dout_h[v * HH + jcol] = hnew;
      }
    }

    // device-wide split barrier across SCAN_NB single-wave blocks
    ++phase;
    if (lane == 0) {
      __hip_atomic_fetch_add(ctr, 1u, __ATOMIC_RELEASE, __HIP_MEMORY_SCOPE_AGENT);
      const unsigned target = (unsigned)SCAN_NB * phase;
      while (__hip_atomic_load(ctr, __ATOMIC_ACQUIRE, __HIP_MEMORY_SCOPE_AGENT) < target)
        __builtin_amdgcn_s_sleep(1);
    }
  }
}

// ---- 5/7. WMMA GEMM: C[M,N] = A_f16[M,HH] x Bpk^T + bias -------------------
// 128 threads = 4 waves. Block tile 64x256; A (64x1024 f16, 128 KB) DMA'd to
// LDS once by the TDM and shared by all waves. Each wave: 4 m-tiles x 4
// n-tiles register tile -> 16 WMMAs per k-tile off 4 LDS A-frags + 4 B-frags.
__global__ void __launch_bounds__(128)
wmma_gemm_kernel(const _Float16* __restrict__ A, const _Float16* __restrict__ Bpk,
                 const float* __restrict__ bias, float* __restrict__ C, int N) {
  extern __shared__ _Float16 asmem[];              // 64 x 1024 f16
  const int tid  = threadIdx.x;
  const int lane = tid & 31;
  const int wv   = tid >> 5;                       // 0..3
  const int m0   = blockIdx.x * 4;                 // first m-tile of block
  const int nt0  = blockIdx.y * 16 + wv * 4;       // wave's 4 n-tiles

  if (wv == 0) {                                   // one TDM descriptor / block
    tdm_load_tile_f16(A + (size_t)m0 * 16 * HH,
                      (unsigned)(size_t)(void*)asmem, 64);
    __builtin_amdgcn_s_wait_tensorcnt(0);
  }
  __syncthreads();

  v8f acc[4][4];
#pragma unroll
  for (int mi = 0; mi < 4; ++mi)
#pragma unroll
    for (int j = 0; j < 4; ++j) acc[mi][j] = v8f{};

  const _Float16* arow = asmem + (size_t)(lane & 15) * HH;
  for (int kt = 0; kt < KT; ++kt) {
    v16h a0 = load_a_frag(arow + 0 * 16 * HH, kt, lane);   // ds_load_b128
    v16h a1 = load_a_frag(arow + 1 * 16 * HH, kt, lane);
    v16h a2 = load_a_frag(arow + 2 * 16 * HH, kt, lane);
    v16h a3 = load_a_frag(arow + 3 * 16 * HH, kt, lane);
#pragma unroll
    for (int j = 0; j < 4; ++j) {
      v16h b = load_b_frag(Bpk, nt0 + j, kt, lane);        // global (L2)
      acc[0][j] = wmma_f16(a0, b, acc[0][j]);
      acc[1][j] = wmma_f16(a1, b, acc[1][j]);
      acc[2][j] = wmma_f16(a2, b, acc[2][j]);
      acc[3][j] = wmma_f16(a3, b, acc[3][j]);
    }
  }

#pragma unroll
  for (int mi = 0; mi < 4; ++mi) {
    const int r0 = (m0 + mi) * 16 + ((lane & 16) ? 8 : 0);
#pragma unroll
    for (int j = 0; j < 4; ++j) {
      int col = (nt0 + j) * 16 + (lane & 15);
      float bb = bias[col];
#pragma unroll
      for (int v = 0; v < 8; ++v)
        C[(size_t)(r0 + v) * N + col] = acc[mi][j][v] + bb;
    }
  }
}

// ---------------------------------------------------------------------------
extern "C" void kernel_launch(void* const* d_in, const int* in_sizes, int n_in,
                              void* d_out, int out_size, void* d_ws, size_t ws_size,
                              hipStream_t stream) {
  (void)in_sizes; (void)n_in; (void)out_size; (void)ws_size;
  const int*   idx    = (const int*)  d_in[0];
  const float* h0     = (const float*)d_in[1];
  const float* w_ih_0 = (const float*)d_in[2];
  const float* w_hh_0 = (const float*)d_in[3];
  const float* b_ih_0 = (const float*)d_in[4];
  const float* b_hh_0 = (const float*)d_in[5];
  const float* w_ih_1 = (const float*)d_in[6];
  const float* w_hh_1 = (const float*)d_in[7];
  const float* b_ih_1 = (const float*)d_in[8];
  const float* b_hh_1 = (const float*)d_in[9];
  const float* head_w = (const float*)d_in[10];
  const float* head_b = (const float*)d_in[11];
  float* out = (float*)d_out;          // logits [B,T,V] ++ h_final [2,B,H]

  char* ws = (char*)d_ws; size_t o = 0;
  auto alloc = [&](size_t bytes) -> void* {
    void* p = ws + o; o = (o + bytes + 255) & ~(size_t)255; return p;
  };
  unsigned* ctr     = (unsigned*)alloc(256);
  float*    gi0     = (float*)   alloc((size_t)BB * TT * G3H * 4);
  float*    gi1     = (float*)   alloc((size_t)BB * TT * G3H * 4);
  _Float16* out0    = (_Float16*)alloc((size_t)BB * TT * HH * 2);
  _Float16* out1    = (_Float16*)alloc((size_t)BB * TT * HH * 2);
  _Float16* pk_whh0 = (_Float16*)alloc((size_t)G3H * HH * 2);
  _Float16* pk_whh1 = (_Float16*)alloc((size_t)G3H * HH * 2);
  _Float16* pk_wih1 = (_Float16*)alloc((size_t)G3H * HH * 2);
  _Float16* pk_head = (_Float16*)alloc((size_t)VV * HH * 2);
  _Float16* h16     = (_Float16*)alloc((size_t)2 * 2 * 16 * HH * 2);
  float*    h32     = (float*)   alloc((size_t)2 * 2 * BB * HH * 4);

  init_kernel<<<dim3(64), dim3(256), 0, stream>>>(h0, h16, h32, ctr);

  auto packN = [&](const float* w, _Float16* pk, int N) {
    size_t total = (size_t)(N / 16) * KT * 32;
    pack_b_kernel<<<dim3((unsigned)((total + 255) / 256)), dim3(256), 0, stream>>>(
        w, pk, N / 16);
  };
  packN(w_hh_0, pk_whh0, G3H);
  packN(w_hh_1, pk_whh1, G3H);
  packN(w_ih_1, pk_wih1, G3H);
  packN(head_w, pk_head, VV);

  gather_gi0_kernel<<<dim3(BB * TT), dim3(256), 0, stream>>>(idx, w_ih_0, b_ih_0, gi0);

  const size_t scan_lds = (size_t)3 * KT * 512 * sizeof(_Float16);   // 96 KB
  const size_t gemm_lds = (size_t)64 * HH * sizeof(_Float16);        // 128 KB

  float* dout_h = out + (size_t)BB * TT * VV;
  gru_scan_kernel<<<dim3(SCAN_NB), dim3(32), scan_lds, stream>>>(
      gi0, pk_whh0, b_hh_0, out0, h16, h32, ctr, dout_h);

  wmma_gemm_kernel<<<dim3((BB * TT) / 64, G3H / 256), dim3(128), gemm_lds, stream>>>(
      out0, pk_wih1, b_ih_1, gi1, G3H);

  gru_scan_kernel<<<dim3(SCAN_NB), dim3(32), scan_lds, stream>>>(
      gi1, pk_whh1, b_hh_1, out1, h16 + (size_t)2 * 16 * HH,
      h32 + (size_t)2 * BB * HH, ctr + 1, dout_h + BB * HH);

  wmma_gemm_kernel<<<dim3((BB * TT) / 64, VV / 256), dim3(128), gemm_lds, stream>>>(
      out1, pk_head, head_b, out, VV);
}
